// DGCRN_67327907332247
// MI455X (gfx1250) — compile-verified, matrix operand order checked by reading, not run on hardware
//
#include <hip/hip_runtime.h>
#include <hip/hip_bf16.h>

typedef __attribute__((ext_vector_type(16))) __bf16 v16bf;
typedef __attribute__((ext_vector_type(8)))  __bf16 v8bf;
typedef __attribute__((ext_vector_type(8)))  float  v8f;

#define B_    32
#define T_    12
#define N_    512
#define HID_  64
#define EMB_  10
#define CP_   80      // padded channels per Chebyshev slot (66 -> 80)
#define KP_   256     // padded GEMM K (3*80 = 240 -> 256)

// Load one WMMA 16-bit fragment: per-lane elements [0..7] = p[0..7],
// elements [8..15] = p[16..23]  (two 16-byte contiguous chunks -> 2x b128).
__device__ __forceinline__ v16bf ld_frag(const __bf16* p) {
  union { v16bf v; v8bf h[2]; } u;
  u.h[0] = *(const v8bf*)(p);
  u.h[1] = *(const v8bf*)(p + 16);
  return u.v;
}

// ---------------------------------------------------------------------------
// Support matrix: softmax(relu(F @ F^T), axis=-1) per row, output bf16.
// ---------------------------------------------------------------------------
__global__ __launch_bounds__(256) void support_kernel(
    const float* __restrict__ feat, size_t featBatchStride,
    __bf16* __restrict__ out) {
  __shared__ float sbuf[N_];
  __shared__ float red[256];
  int rowId = blockIdx.x;
  int b = rowId >> 9;
  int n = rowId & (N_ - 1);
  const float* fb = feat + (size_t)b * featBatchStride;
  float fn[EMB_];
#pragma unroll
  for (int e = 0; e < EMB_; ++e) fn[e] = fb[(size_t)n * EMB_ + e];
  int tid = threadIdx.x;
  for (int m = tid; m < N_; m += 256) {
    const float* fm = fb + (size_t)m * EMB_;
    float s = 0.f;
#pragma unroll
    for (int e = 0; e < EMB_; ++e) s += fn[e] * fm[e];
    sbuf[m] = s > 0.f ? s : 0.f;
  }
  __syncthreads();
  float mx = fmaxf(sbuf[tid], sbuf[tid + 256]);
  red[tid] = mx;
  __syncthreads();
  for (int off = 128; off > 0; off >>= 1) {
    if (tid < off) red[tid] = fmaxf(red[tid], red[tid + off]);
    __syncthreads();
  }
  mx = red[0];
  __syncthreads();
  float sum = __expf(sbuf[tid] - mx) + __expf(sbuf[tid + 256] - mx);
  red[tid] = sum;
  __syncthreads();
  for (int off = 128; off > 0; off >>= 1) {
    if (tid < off) red[tid] += red[tid + off];
    __syncthreads();
  }
  float inv = 1.f / red[0];
  __bf16* ob = out + (size_t)rowId * N_;
  for (int m = tid; m < N_; m += 256) {
    ob[m] = (__bf16)(__expf(sbuf[m] - mx) * inv);
  }
}

// ---------------------------------------------------------------------------
// Repack f32 weight (198 x C) into TRANSPOSED padded bf16 Wt (C x 256).
// ---------------------------------------------------------------------------
__global__ __launch_bounds__(256) void pack_w_kernel(
    const float* __restrict__ src, __bf16* __restrict__ dst, int C) {
  int idx = blockIdx.x * 256 + threadIdx.x;   // KP_ * C total
  int kp = idx / C;
  int c  = idx % C;
  float v = 0.f;
  if (kp < 3 * CP_) {
    int slot = kp / CP_, within = kp % CP_;
    if (within < 66) v = src[(size_t)(slot * 66 + within) * C + c];
  }
  dst[(size_t)c * KP_ + kp] = (__bf16)v;
}

// ---------------------------------------------------------------------------
// Pack the 66 input channels into slot0 of BOTH layouts:
//   XGr: (b*512+n) x 256     (node-major, GEMM A operand)
//   XGt: per batch 256 x 512 (channel-major, graph-prop B operand)
// mode 0: enc inp | 1: enc cand (z*h) | 2: dec inp | 3: dec cand (z*h)
// ---------------------------------------------------------------------------
__global__ __launch_bounds__(256) void pack_kernel(
    const float* __restrict__ x, const float* __restrict__ ycov,
    const float* __restrict__ go, const float* __restrict__ h,
    const float* __restrict__ zr, __bf16* __restrict__ XGr,
    __bf16* __restrict__ XGt, int t, int mode) {
  int idx = blockIdx.x * 256 + threadIdx.x;   // B_*N_*CP_
  int c   = idx % CP_;
  int row = idx / CP_;                        // b*512 + n
  int b = row >> 9;
  int n = row & (N_ - 1);
  float v = 0.f;
  if (c < 2) {
    if (mode < 2) {
      v = x[(((size_t)b * T_ + t) * N_ + n) * 2 + c];
    } else {
      v = (c == 0) ? go[row] : ycov[((size_t)b * T_ + t) * N_ + n];
    }
  } else if (c < 66) {
    float hv = h[(size_t)row * HID_ + (c - 2)];
    if (mode & 1) hv *= zr[(size_t)row * 128 + (c - 2)];   // z gate
    v = hv;
  }
  __bf16 bv = (__bf16)v;
  XGr[(size_t)row * KP_ + c] = bv;
  XGt[((size_t)b * KP_ + c) * N_ + n] = bv;
}

// ---------------------------------------------------------------------------
// Graph propagation: Y[b] = alpha * S[b] @ X[b] + beta * Base[b]
// One wave owns a 16-node row strip and ALL 5 channel tiles: the S-row
// fragment is loaded once per K-step and reused for 5 WMMAs.
// ---------------------------------------------------------------------------
__global__ __launch_bounds__(256) void graph_prop_kernel(
    const __bf16* __restrict__ S, size_t sBatchStride,
    const __bf16* __restrict__ Xt, const __bf16* __restrict__ BaseT,
    __bf16* __restrict__ Yr, __bf16* __restrict__ Yt,
    float alpha, float beta) {
  int wave = blockIdx.x * 8 + (threadIdx.x >> 5);  // b*32 + mt, 1024 waves
  int lane = threadIdx.x & 31;
  int b  = wave >> 5;
  int mt = wave & 31;
  int m  = lane & 15;
  int kb = (lane < 16) ? 0 : 8;
  const __bf16* arow  = S  + (size_t)b * sBatchStride + (size_t)(mt * 16 + m) * N_;
  const __bf16* bbase = Xt + (size_t)b * KP_ * N_ + (size_t)m * N_;
  v8f acc[5] = {};
  for (int k0 = 0; k0 < N_; k0 += 32) {
    v16bf a = ld_frag(arow + k0 + kb);
    if (k0 + 32 < N_) __builtin_prefetch(arow + k0 + 32 + kb, 0, 3);
#pragma unroll
    for (int ct = 0; ct < 5; ++ct) {
      v16bf bm = ld_frag(bbase + (size_t)(ct * 16) * N_ + k0 + kb);
      acc[ct] = __builtin_amdgcn_wmma_f32_16x16x32_bf16(
          false, a, false, bm, (short)0, acc[ct], false, false);
    }
  }
  int rbase = mt * 16 + ((lane < 16) ? 0 : 8);
#pragma unroll
  for (int ct = 0; ct < 5; ++ct) {
    int ncol = ct * 16 + m;                    // local channel (0..79)
    float vals[8];
    if (BaseT) {
      v8bf bb = *(const v8bf*)(BaseT + ((size_t)b * KP_ + ncol) * N_ + rbase);
#pragma unroll
      for (int r = 0; r < 8; ++r) vals[r] = acc[ct][r] * alpha + beta * (float)bb[r];
    } else {
#pragma unroll
      for (int r = 0; r < 8; ++r) vals[r] = acc[ct][r] * alpha;
    }
    v8bf ov;
#pragma unroll
    for (int r = 0; r < 8; ++r) ov[r] = (__bf16)vals[r];
    // channel-major store: fixed channel row, 8 consecutive nodes -> b128
    *(v8bf*)(Yt + ((size_t)b * KP_ + ncol) * N_ + rbase) = ov;
    // node-major store: coalesced across lanes
    __bf16* yr = Yr + ((size_t)b * N_ + rbase) * KP_ + ncol;
#pragma unroll
    for (int r = 0; r < 8; ++r) yr[(size_t)r * KP_] = ov[r];
  }
}

// ---------------------------------------------------------------------------
// Gate GEMM: zr = sigmoid(XGr @ W + b). One wave owns a 16-row strip and
// all 8 column tiles (A fragment reused 8x per K-step).
// ---------------------------------------------------------------------------
__global__ __launch_bounds__(256) void gemm_gate_kernel(
    const __bf16* __restrict__ XGr, const __bf16* __restrict__ Wt,
    const float* __restrict__ bias, float* __restrict__ zr) {
  const int NC = 128;
  int mt   = blockIdx.x * 8 + (threadIdx.x >> 5);   // 1024 row tiles
  int lane = threadIdx.x & 31;
  int m  = lane & 15;
  int kb = (lane < 16) ? 0 : 8;
  const __bf16* arow = XGr + (size_t)(mt * 16 + m) * KP_;
  const __bf16* brow = Wt + (size_t)m * KP_;
  v8f acc[8] = {};
#pragma unroll
  for (int k0 = 0; k0 < KP_; k0 += 32) {
    v16bf a = ld_frag(arow + k0 + kb);
#pragma unroll
    for (int nt = 0; nt < 8; ++nt) {
      v16bf bm = ld_frag(brow + (size_t)(nt * 16) * KP_ + k0 + kb);
      acc[nt] = __builtin_amdgcn_wmma_f32_16x16x32_bf16(
          false, a, false, bm, (short)0, acc[nt], false, false);
    }
  }
  int rbase = mt * 16 + ((lane < 16) ? 0 : 8);
#pragma unroll
  for (int nt = 0; nt < 8; ++nt) {
    int ncol = nt * 16 + m;
    float bv = bias[ncol];
#pragma unroll
    for (int r = 0; r < 8; ++r) {
      float v = acc[nt][r] + bv;
      zr[(size_t)(rbase + r) * NC + ncol] = 1.f / (1.f + __expf(-v));
    }
  }
}

// ---------------------------------------------------------------------------
// Update GEMM + GRU update: hc = tanh(XGr @ W + b); h = r*h + (1-r)*hc.
// One wave owns a 16-row strip and all 4 column tiles.
// ---------------------------------------------------------------------------
__global__ __launch_bounds__(256) void gemm_update_kernel(
    const __bf16* __restrict__ XGr, const __bf16* __restrict__ Wt,
    const float* __restrict__ bias, const float* __restrict__ zr,
    float* __restrict__ h) {
  const int NC = 64;
  int mt   = blockIdx.x * 8 + (threadIdx.x >> 5);   // 1024 row tiles
  int lane = threadIdx.x & 31;
  int m  = lane & 15;
  int kb = (lane < 16) ? 0 : 8;
  const __bf16* arow = XGr + (size_t)(mt * 16 + m) * KP_;
  const __bf16* brow = Wt + (size_t)m * KP_;
  v8f acc[4] = {};
#pragma unroll
  for (int k0 = 0; k0 < KP_; k0 += 32) {
    v16bf a = ld_frag(arow + k0 + kb);
#pragma unroll
    for (int nt = 0; nt < 4; ++nt) {
      v16bf bm = ld_frag(brow + (size_t)(nt * 16) * KP_ + k0 + kb);
      acc[nt] = __builtin_amdgcn_wmma_f32_16x16x32_bf16(
          false, a, false, bm, (short)0, acc[nt], false, false);
    }
  }
  int rbase = mt * 16 + ((lane < 16) ? 0 : 8);
#pragma unroll
  for (int nt = 0; nt < 4; ++nt) {
    int ncol = nt * 16 + m;
    float bv = bias[ncol];
#pragma unroll
    for (int r = 0; r < 8; ++r) {
      int row = rbase + r;
      float hc = tanhf(acc[nt][r] + bv);
      float rv = zr[(size_t)row * 128 + 64 + ncol];
      float ho = h[(size_t)row * NC + ncol];
      h[(size_t)row * NC + ncol] = rv * ho + (1.f - rv) * hc;
    }
  }
}

// ---------------------------------------------------------------------------
__global__ __launch_bounds__(256) void ne_kernel(
    const float* __restrict__ h, const float* __restrict__ hW,
    const float* __restrict__ hb, float* __restrict__ ne) {
  int row = blockIdx.x * 256 + threadIdx.x;   // B_*N_
  float hv[HID_];
#pragma unroll
  for (int i = 0; i < HID_; ++i) hv[i] = h[(size_t)row * HID_ + i];
#pragma unroll
  for (int e = 0; e < EMB_; ++e) {
    float s = hb[e];
    for (int i = 0; i < HID_; ++i) s += hv[i] * hW[(size_t)i * EMB_ + e];
    ne[(size_t)row * EMB_ + e] = s;
  }
}

// ---------------------------------------------------------------------------
__global__ __launch_bounds__(256) void proj_kernel(
    const float* __restrict__ h, const float* __restrict__ pW,
    const float* __restrict__ pb, float* __restrict__ go,
    float* __restrict__ out, int t) {
  int row = blockIdx.x * 256 + threadIdx.x;   // B_*N_
  int b = row >> 9;
  int n = row & (N_ - 1);
  float s = pb[0];
  for (int i = 0; i < HID_; ++i) s += h[(size_t)row * HID_ + i] * pW[i];
  go[row] = s;
  out[((size_t)b * T_ + t) * N_ + n] = s;
}

// ---------------------------------------------------------------------------
extern "C" void kernel_launch(void* const* d_in, const int* in_sizes, int n_in,
                              void* d_out, int out_size, void* d_ws, size_t ws_size,
                              hipStream_t stream) {
  (void)in_sizes; (void)n_in; (void)out_size; (void)ws_size;
  const float* x      = (const float*)d_in[0];
  const float* ycov   = (const float*)d_in[1];
  const float* nemb   = (const float*)d_in[2];
  const float* enc_gW = (const float*)d_in[3];
  const float* enc_gb = (const float*)d_in[4];
  const float* enc_uW = (const float*)d_in[5];
  const float* enc_ub = (const float*)d_in[6];
  const float* dec_gW = (const float*)d_in[7];
  const float* dec_gb = (const float*)d_in[8];
  const float* dec_uW = (const float*)d_in[9];
  const float* dec_ub = (const float*)d_in[10];
  const float* proj_W = (const float*)d_in[11];
  const float* proj_b = (const float*)d_in[12];
  const float* hyp_W  = (const float*)d_in[13];
  const float* hyp_b  = (const float*)d_in[14];
  float* out = (float*)d_out;

  char* w = (char*)d_ws;
  size_t off = 0;
  auto alloc = [&](size_t bytes) -> void* {
    void* p = w + off;
    off += (bytes + 255) & ~(size_t)255;
    return p;
  };
  __bf16* sup_en = (__bf16*)alloc((size_t)N_ * N_ * 2);
  __bf16* sup_de = (__bf16*)alloc((size_t)B_ * N_ * N_ * 2);
  float*  h      = (float*) alloc((size_t)B_ * N_ * HID_ * 4);
  float*  zr     = (float*) alloc((size_t)B_ * N_ * 128 * 4);
  __bf16* XGr    = (__bf16*)alloc((size_t)B_ * N_ * KP_ * 2);  // node-major
  __bf16* XGt    = (__bf16*)alloc((size_t)B_ * KP_ * N_ * 2);  // channel-major
  float*  go     = (float*) alloc((size_t)B_ * N_ * 4);
  float*  ne     = (float*) alloc((size_t)B_ * N_ * EMB_ * 4);
  __bf16* gWpE   = (__bf16*)alloc((size_t)KP_ * 128 * 2);      // transposed
  __bf16* uWpE   = (__bf16*)alloc((size_t)KP_ * 64 * 2);
  __bf16* gWpD   = (__bf16*)alloc((size_t)KP_ * 128 * 2);
  __bf16* uWpD   = (__bf16*)alloc((size_t)KP_ * 64 * 2);

  // zero workspace: h0=0, go0=0, zero pad channels in XGr/XGt
  hipMemsetAsync(d_ws, 0, off, stream);

  pack_w_kernel<<<128, 256, 0, stream>>>(enc_gW, gWpE, 128);
  pack_w_kernel<<< 64, 256, 0, stream>>>(enc_uW, uWpE, 64);
  pack_w_kernel<<<128, 256, 0, stream>>>(dec_gW, gWpD, 128);
  pack_w_kernel<<< 64, 256, 0, stream>>>(dec_uW, uWpD, 64);

  support_kernel<<<N_, 256, 0, stream>>>(nemb, 0, sup_en);

  const int PACK_BLK = (B_ * N_ * CP_) / 256;     // 5120
  const int PROP_BLK = (B_ * 32) / 8;             // 128 (1024 waves)
  const int GATE_BLK = (B_ * N_ / 16) / 8;        // 128 (1024 waves)
  const int UPD_BLK  = (B_ * N_ / 16) / 8;        // 128 (1024 waves)

  // slot pointers: XGr offsets by channel; XGt offsets by channel*N_
  __bf16* XGt0 = XGt;
  __bf16* XGt1 = XGt + (size_t)CP_ * N_;
  __bf16* XGt2 = XGt + (size_t)2 * CP_ * N_;

  // ---------------- encoder ----------------
  for (int t = 0; t < T_; ++t) {
    pack_kernel<<<PACK_BLK, 256, 0, stream>>>(x, ycov, go, h, zr, XGr, XGt, t, 0);
    graph_prop_kernel<<<PROP_BLK, 256, 0, stream>>>(sup_en, 0, XGt0, nullptr, XGr + CP_,     XGt1, 1.f,  0.f);
    graph_prop_kernel<<<PROP_BLK, 256, 0, stream>>>(sup_en, 0, XGt1, XGt0,    XGr + 2 * CP_, XGt2, 2.f, -1.f);
    gemm_gate_kernel<<<GATE_BLK, 256, 0, stream>>>(XGr, gWpE, enc_gb, zr);
    pack_kernel<<<PACK_BLK, 256, 0, stream>>>(x, ycov, go, h, zr, XGr, XGt, t, 1);
    graph_prop_kernel<<<PROP_BLK, 256, 0, stream>>>(sup_en, 0, XGt0, nullptr, XGr + CP_,     XGt1, 1.f,  0.f);
    graph_prop_kernel<<<PROP_BLK, 256, 0, stream>>>(sup_en, 0, XGt1, XGt0,    XGr + 2 * CP_, XGt2, 2.f, -1.f);
    gemm_update_kernel<<<UPD_BLK, 256, 0, stream>>>(XGr, uWpE, enc_ub, zr, h);
  }

  // ---------------- hyper support ----------------
  ne_kernel<<<(B_ * N_) / 256, 256, 0, stream>>>(h, hyp_W, hyp_b, ne);
  support_kernel<<<B_ * N_, 256, 0, stream>>>(ne, (size_t)N_ * EMB_, sup_de);

  // ---------------- decoder ----------------
  const size_t SDE = (size_t)N_ * N_;
  for (int t = 0; t < T_; ++t) {
    pack_kernel<<<PACK_BLK, 256, 0, stream>>>(x, ycov, go, h, zr, XGr, XGt, t, 2);
    graph_prop_kernel<<<PROP_BLK, 256, 0, stream>>>(sup_de, SDE, XGt0, nullptr, XGr + CP_,     XGt1, 1.f,  0.f);
    graph_prop_kernel<<<PROP_BLK, 256, 0, stream>>>(sup_de, SDE, XGt1, XGt0,    XGr + 2 * CP_, XGt2, 2.f, -1.f);
    gemm_gate_kernel<<<GATE_BLK, 256, 0, stream>>>(XGr, gWpD, dec_gb, zr);
    pack_kernel<<<PACK_BLK, 256, 0, stream>>>(x, ycov, go, h, zr, XGr, XGt, t, 3);
    graph_prop_kernel<<<PROP_BLK, 256, 0, stream>>>(sup_de, SDE, XGt0, nullptr, XGr + CP_,     XGt1, 1.f,  0.f);
    graph_prop_kernel<<<PROP_BLK, 256, 0, stream>>>(sup_de, SDE, XGt1, XGt0,    XGr + 2 * CP_, XGt2, 2.f, -1.f);
    gemm_update_kernel<<<UPD_BLK, 256, 0, stream>>>(XGr, uWpD, dec_ub, zr, h);
    proj_kernel<<<(B_ * N_) / 256, 256, 0, stream>>>(h, proj_W, proj_b, go, out, t);
  }
}
